// GATConvNN_8693013807643
// MI455X (gfx1250) — compile-verified, matrix operand order checked by reading, not run on hardware
//
#include <hip/hip_runtime.h>
#include <hip/hip_bf16.h>
#include <math.h>

#define NN_ 50000
#define E_  800000
#define E2_ (E_ + NN_)
#define HID_ 128
#define FIN_ 30      // F_IN * INPUT_SIZE
#define NOUT_ 10     // F_OUT * FH
#define NLAYERS_ 5
#define NEG_SLOPE_ 0.2f
#define EPS_ 1e-5f

typedef __attribute__((ext_vector_type(16))) __bf16 v16bf;
typedef __attribute__((ext_vector_type(8)))  float  v8f;

union V16U { uint4 u[2]; v16bf v; };

__device__ __forceinline__ unsigned short f32_to_bf16(float f) {
  unsigned int u = __float_as_uint(f);
  u = u + 0x7FFFu + ((u >> 16) & 1u);   // round-to-nearest-even
  return (unsigned short)(u >> 16);
}
__device__ __forceinline__ unsigned int fenc(float f) {
  unsigned int u = __float_as_uint(f);
  return (u & 0x80000000u) ? ~u : (u | 0x80000000u);
}
__device__ __forceinline__ float fdec(unsigned int e) {
  return __uint_as_float((e & 0x80000000u) ? (e & 0x7FFFFFFFu) : ~e);
}
__device__ __forceinline__ int clampi(int v, int hi) {
  return v < 0 ? 0 : (v >= hi ? hi - 1 : v);
}

// ---------------- fills ----------------
__global__ void k_fill_f32(float* p, long n, float v) {
  long i = (long)blockIdx.x * blockDim.x + threadIdx.x;
  if (i < n) p[i] = v;
}
__global__ void k_fill_u32(unsigned* p, long n, unsigned v) {
  long i = (long)blockIdx.x * blockDim.x + threadIdx.x;
  if (i < n) p[i] = v;
}

// ---------------- embed + layernorm + relu ----------------
__global__ void k_embed(const float* __restrict__ x, const float* __restrict__ W,
                        const float* __restrict__ b, const float* __restrict__ g,
                        const float* __restrict__ bb,
                        float* __restrict__ h, unsigned short* __restrict__ hbf) {
  int n = blockIdx.x, j = threadIdx.x;     // 128 threads/node
  const float* xr = x + (size_t)n * FIN_;
  float acc = b[j];
  #pragma unroll
  for (int i = 0; i < FIN_; ++i) acc += xr[i] * W[i * HID_ + j];
  __shared__ float red[HID_];
  red[j] = acc; __syncthreads();
  for (int s = 64; s > 0; s >>= 1) { if (j < s) red[j] += red[j + s]; __syncthreads(); }
  float mu = red[0] * (1.0f / HID_);
  __syncthreads();
  float d = acc - mu;
  red[j] = d * d; __syncthreads();
  for (int s = 64; s > 0; s >>= 1) { if (j < s) red[j] += red[j + s]; __syncthreads(); }
  float var = red[0] * (1.0f / HID_);
  float v = d * rsqrtf(var + EPS_) * g[j] + bb[j];
  v = fmaxf(v, 0.0f);
  h[(size_t)n * HID_ + j] = v;
  hbf[(size_t)n * HID_ + j] = f32_to_bf16(v);
}

// ---------------- self-loop attr (mean of incoming) ----------------
__global__ void k_deg_attr(const int* __restrict__ ei, const float* __restrict__ ea,
                           float* __restrict__ deg, float* __restrict__ la) {
  int e = blockIdx.x * blockDim.x + threadIdx.x;
  if (e >= E_) return;
  int d = clampi(ei[E_ + e], NN_);
  atomicAdd(&deg[d], 1.0f);
  atomicAdd(&la[d * 3 + 0], ea[(size_t)e * 3 + 0]);
  atomicAdd(&la[d * 3 + 1], ea[(size_t)e * 3 + 1]);
  atomicAdd(&la[d * 3 + 2], ea[(size_t)e * 3 + 2]);
}
__global__ void k_norm_attr(float* la, const float* deg) {
  int n = blockIdx.x * blockDim.x + threadIdx.x;
  if (n >= NN_) return;
  float dg = fmaxf(deg[n], 1.0f);
  la[n * 3 + 0] /= dg; la[n * 3 + 1] /= dg; la[n * 3 + 2] /= dg;
}

// ---------------- pack W (128x128 f32 row-major) into WMMA-B lane order ----------------
// layout: [nt(8)][kt(4)][lane(32)][16 bf16], lane<16 -> K base..+7,+16..+23 ; lane>=16 -> +8..
__global__ void k_packW(const float* __restrict__ W, unsigned short* __restrict__ Wp) {
  int t = blockIdx.x * blockDim.x + threadIdx.x;
  if (t >= HID_ * HID_) return;
  int j = t & 15, lane = (t >> 4) & 31, kt = (t >> 9) & 3, nt = t >> 11;
  int half = lane >> 4, n = nt * 16 + (lane & 15);
  int k = kt * 32 + half * 8 + (j < 8 ? j : j + 8);
  Wp[t] = f32_to_bf16(W[k * HID_ + n]);
}

// ---------------- GEMM: hp[N,128] = h_bf16[N,128] @ Wp (WMMA bf16 -> f32) ----------------
__global__ void __launch_bounds__(128) k_gemm(const unsigned short* __restrict__ A,
                                              const unsigned short* __restrict__ Wp,
                                              float* __restrict__ C) {
  int wave = (blockIdx.x * blockDim.x + threadIdx.x) >> 5;
  int lane = threadIdx.x & 31;
  const int tiles = NN_ / 16;            // 3125, exact
  if (wave >= tiles) return;             // wave-uniform exit: EXEC all-ones inside
  int row0 = wave * 16;
  int half = lane >> 4, m = lane & 15;

  v8f acc[8];
  #pragma unroll
  for (int nt = 0; nt < 8; ++nt)
    #pragma unroll
    for (int i = 0; i < 8; ++i) acc[nt][i] = 0.0f;

  #pragma unroll
  for (int kt = 0; kt < 4; ++kt) {
    V16U a;
    const uint4* pa = (const uint4*)(A + (size_t)(row0 + m) * HID_ + kt * 32 + half * 8);
    a.u[0] = pa[0];      // K = base .. base+7
    a.u[1] = pa[2];      // K = base+16 .. base+23
    #pragma unroll
    for (int nt = 0; nt < 8; ++nt) {
      V16U b;
      const uint4* pb = (const uint4*)(Wp + ((size_t)(nt * 4 + kt) * 32 + lane) * 16);
      b.u[0] = pb[0]; b.u[1] = pb[1];
      acc[nt] = __builtin_amdgcn_wmma_f32_16x16x32_bf16(
          false, a.v, false, b.v, (short)0, acc[nt], false, false);
    }
  }
  #pragma unroll
  for (int nt = 0; nt < 8; ++nt)
    #pragma unroll
    for (int r = 0; r < 8; ++r)
      C[(size_t)(row0 + half * 8 + r) * HID_ + nt * 16 + m] = acc[nt][r];
}

// ---------------- per-node attention dots: as = hp@a_src, ad = hp@a_dst ----------------
__global__ void k_nodedots(const float* __restrict__ hp, const float* __restrict__ asrc,
                           const float* __restrict__ adst,
                           float* __restrict__ as_, float* __restrict__ ad_) {
  int tid = blockIdx.x * blockDim.x + threadIdx.x;
  int n = tid >> 5, lane = tid & 31;
  if (n >= NN_) return;
  float s1 = 0.f, s2 = 0.f;
  #pragma unroll
  for (int r = 0; r < 4; ++r) {
    float v = hp[(size_t)n * HID_ + lane + r * 32];
    s1 += v * asrc[lane + r * 32];
    s2 += v * adst[lane + r * 32];
  }
  #pragma unroll
  for (int off = 16; off > 0; off >>= 1) {
    s1 += __shfl_xor(s1, off, 32);
    s2 += __shfl_xor(s2, off, 32);
  }
  if (lane == 0) { as_[n] = s1; ad_[n] = s2; }
}

// ---------------- wa[d] = We[d] . a_e  (edge path collapsed to 3 scalars) ----------------
__global__ void k_wa(const float* __restrict__ We, const float* __restrict__ ae,
                     float* __restrict__ wa) {
  int lane = threadIdx.x;                 // 32 threads
  #pragma unroll
  for (int d = 0; d < 3; ++d) {
    float s = 0.f;
    #pragma unroll
    for (int r = 0; r < 4; ++r) s += We[d * HID_ + lane + r * 32] * ae[lane + r * 32];
    #pragma unroll
    for (int off = 16; off > 0; off >>= 1) s += __shfl_xor(s, off, 32);
    if (lane == 0) wa[d] = s;
  }
}

// ---------------- per-edge alpha + leaky relu + segment max ----------------
__global__ void k_alpha(const int* __restrict__ ei, const float* __restrict__ ea,
                        const float* __restrict__ la, const float* __restrict__ as_,
                        const float* __restrict__ ad_, const float* __restrict__ wa,
                        float* __restrict__ alpha, unsigned* __restrict__ amax) {
  int e = blockIdx.x * blockDim.x + threadIdx.x;
  if (e >= E2_) return;
  int s, d; const float* a;
  if (e < E_) { s = clampi(ei[e], NN_); d = clampi(ei[E_ + e], NN_); a = ea + (size_t)e * 3; }
  else        { s = d = e - E_;                                       a = la + (size_t)s * 3; }
  float al = as_[s] + ad_[d] + a[0] * wa[0] + a[1] * wa[1] + a[2] * wa[2];
  al = al >= 0.f ? al : NEG_SLOPE_ * al;
  alpha[e] = al;
  atomicMax(amax + d, fenc(al));
}

// ---------------- ex = exp(alpha - amax[dst]); denom = segment_sum(ex) ----------------
__global__ void k_exp(const int* __restrict__ ei, float* __restrict__ alpha,
                      const unsigned* __restrict__ amax, float* __restrict__ denom) {
  int e = blockIdx.x * blockDim.x + threadIdx.x;
  if (e >= E2_) return;
  int d = (e < E_) ? clampi(ei[E_ + e], NN_) : e - E_;
  float ex = __expf(alpha[e] - fdec(amax[d]));
  alpha[e] = ex;
  atomicAdd(&denom[d], ex);
}

// ---------------- message aggregation: one wave per edge, float4 per lane ----------------
__global__ void k_agg(const int* __restrict__ ei, const float* __restrict__ ex,
                      const float* __restrict__ denom, const float* __restrict__ hp,
                      float* __restrict__ agg) {
  int tid = blockIdx.x * blockDim.x + threadIdx.x;
  int e = tid >> 5, lane = tid & 31;
  if (e >= E2_) return;
  int s, d;
  if (e < E_) { s = clampi(ei[e], NN_); d = clampi(ei[E_ + e], NN_); }
  else        { s = d = e - E_; }
  float w = ex[e] / denom[d];
  float4 hv = ((const float4*)(hp + (size_t)s * HID_))[lane];
  float* o = agg + (size_t)d * HID_ + lane * 4;
  atomicAdd(o + 0, hv.x * w);
  atomicAdd(o + 1, hv.y * w);
  atomicAdd(o + 2, hv.z * w);
  atomicAdd(o + 3, hv.w * w);
}

// ---------------- h = relu(agg + bias); also refresh bf16 copy ----------------
__global__ void k_final(const float* __restrict__ agg, const float* __restrict__ b,
                        float* __restrict__ h, unsigned short* __restrict__ hbf) {
  long i = (long)blockIdx.x * blockDim.x + threadIdx.x;
  if (i >= (long)NN_ * HID_) return;
  int j = (int)(i & (HID_ - 1));
  float v = fmaxf(agg[i] + b[j], 0.f);
  h[i] = v;
  hbf[i] = f32_to_bf16(v);
}

// ---------------- decoder ----------------
__global__ void k_dec(const float* __restrict__ h, const float* __restrict__ W,
                      const float* __restrict__ b, float* __restrict__ y) {
  long t = (long)blockIdx.x * blockDim.x + threadIdx.x;
  if (t >= (long)NN_ * NOUT_) return;
  int n = (int)(t / NOUT_), o = (int)(t % NOUT_);
  float s = b[o];
  const float* hr = h + (size_t)n * HID_;
  #pragma unroll 8
  for (int k = 0; k < HID_; ++k) s += hr[k] * W[k * NOUT_ + o];
  y[t] = s;
}

extern "C" void kernel_launch(void* const* d_in, const int* in_sizes, int n_in,
                              void* d_out, int out_size, void* d_ws, size_t ws_size,
                              hipStream_t stream) {
  (void)in_sizes; (void)n_in; (void)out_size; (void)ws_size;
  const float* x    = (const float*)d_in[0];
  const int*   ei   = (const int*)  d_in[1];
  const float* ea   = (const float*)d_in[2];
  const float* embW = (const float*)d_in[3];
  const float* embB = (const float*)d_in[4];
  const float* lng  = (const float*)d_in[5];
  const float* lnb  = (const float*)d_in[6];
  const float* gatW = (const float*)d_in[7];
  const float* gatB = (const float*)d_in[8];
  const float* attS = (const float*)d_in[9];
  const float* attD = (const float*)d_in[10];
  const float* edgW = (const float*)d_in[11];
  const float* attE = (const float*)d_in[12];
  const float* decW = (const float*)d_in[13];
  const float* decB = (const float*)d_in[14];
  float* out = (float*)d_out;

  char* p = (char*)d_ws;
  auto carve = [&](size_t bytes) -> char* {
    char* r = p; p += (bytes + 255) & ~(size_t)255; return r;
  };
  float*          h     = (float*)carve((size_t)NN_ * HID_ * 4);
  float*          hp    = (float*)carve((size_t)NN_ * HID_ * 4);
  float*          agg   = (float*)carve((size_t)NN_ * HID_ * 4);
  unsigned short* hbf   = (unsigned short*)carve((size_t)NN_ * HID_ * 2);
  unsigned short* Wp    = (unsigned short*)carve((size_t)HID_ * HID_ * 2);
  float*          as_   = (float*)carve((size_t)NN_ * 4);
  float*          ad_   = (float*)carve((size_t)NN_ * 4);
  float*          alpha = (float*)carve((size_t)E2_ * 4);
  unsigned*       amax  = (unsigned*)carve((size_t)NN_ * 4);
  float*          denom = (float*)carve((size_t)NN_ * 4);
  float*          deg   = (float*)carve((size_t)NN_ * 4);
  float*          la    = (float*)carve((size_t)NN_ * 3 * 4);
  float*          wa    = (float*)carve(64);

  // 1) embedding + layernorm + relu
  k_embed<<<NN_, HID_, 0, stream>>>(x, embW, embB, lng, lnb, h, hbf);

  // 2) self-loop attr = per-dst mean of incoming edge_attr
  long nla = (long)NN_ * 3;
  k_fill_f32<<<(NN_ + 255) / 256, 256, 0, stream>>>(deg, NN_, 0.f);
  k_fill_f32<<<(int)((nla + 255) / 256), 256, 0, stream>>>(la, nla, 0.f);
  k_deg_attr<<<(E_ + 255) / 256, 256, 0, stream>>>(ei, ea, deg, la);
  k_norm_attr<<<(NN_ + 255) / 256, 256, 0, stream>>>(la, deg);

  const long nh = (long)NN_ * HID_;
  for (int l = 0; l < NLAYERS_; ++l) {
    k_packW<<<(HID_ * HID_ + 255) / 256, 256, 0, stream>>>(gatW + (size_t)l * HID_ * HID_, Wp);
    int tiles = NN_ / 16;
    k_gemm<<<(tiles + 3) / 4, 128, 0, stream>>>(hbf, Wp, hp);
    k_nodedots<<<(NN_ * 32 + 127) / 128, 128, 0, stream>>>(hp, attS + l * HID_, attD + l * HID_, as_, ad_);
    k_wa<<<1, 32, 0, stream>>>(edgW + (size_t)l * 3 * HID_, attE + l * HID_, wa);
    k_fill_u32<<<(NN_ + 255) / 256, 256, 0, stream>>>(amax, NN_, 0x007FFFFFu); // fenc(-inf)
    k_fill_f32<<<(NN_ + 255) / 256, 256, 0, stream>>>(denom, NN_, 0.f);
    k_fill_f32<<<(int)((nh + 255) / 256), 256, 0, stream>>>(agg, nh, 0.f);
    k_alpha<<<(E2_ + 255) / 256, 256, 0, stream>>>(ei, ea, la, as_, ad_, wa, alpha, amax);
    k_exp<<<(E2_ + 255) / 256, 256, 0, stream>>>(ei, alpha, amax, denom);
    long tt = (long)E2_ * 32;
    k_agg<<<(int)((tt + 255) / 256), 256, 0, stream>>>(ei, alpha, denom, hp, agg);
    k_final<<<(int)((nh + 255) / 256), 256, 0, stream>>>(agg, gatB + l * HID_, h, hbf);
  }

  long td = (long)NN_ * NOUT_;
  k_dec<<<(int)((td + 255) / 256), 256, 0, stream>>>(h, decW, decB, out);
}
